// MultiKernelSelfAttention_5531917877537
// MI455X (gfx1250) — matrix operand order repaired and checked by reading, not compiled, not run on hardware
//
#include <hip/hip_runtime.h>
#include <hip/hip_bf16.h>
#include <stdint.h>

typedef _Float16 h16;
typedef __attribute__((ext_vector_type(16))) _Float16 v16h;
typedef __attribute__((ext_vector_type(8)))  _Float16 v8h;
typedef __attribute__((ext_vector_type(8)))  float    v8f;
typedef __attribute__((ext_vector_type(4)))  int      v4i;

#define B_    4
#define C_    256
#define Hh_   56
#define Wd_   56
#define HW_   3136
#define RC_   32
#define NB_   4          // q branches
#define NT_I  196        // HW / 16
#define JST   98         // HW / 32
#define FOURC 1024       // 4*C
#define NPART 784        // B * NT_I

// ---- optional gfx1250 async LDS pipeline (probe-guarded) -------------------
#if defined(__has_builtin)
#if __has_builtin(__builtin_amdgcn_global_load_async_to_lds_b128) && \
    __has_builtin(__builtin_amdgcn_s_wait_asynccnt)
#define HAVE_ASYNC_LDS 1
#endif
#endif
#ifdef HAVE_ASYNC_LDS
typedef __attribute__((address_space(1))) v4i v4i_g;   // global int4
typedef __attribute__((address_space(3))) v4i v4i_l;   // LDS int4
#define AS1(p) ((v4i_g*)(uintptr_t)(p))
#define AS3(p) ((v4i_l*)(unsigned)(uintptr_t)(p))
#endif

#define VSHUF16(a, b) __builtin_shufflevector(a, b, 0,1,2,3,4,5,6,7,8,9,10,11,12,13,14,15)

// ---------------------------------------------------------------------------
// x [B][C][HW] f32  ->  x_t [B][HW][C] f16  (channel-contiguous for B-frags)
// ---------------------------------------------------------------------------
__global__ void mksa_xt(const float* __restrict__ x, h16* __restrict__ xt) {
    int idx = blockIdx.x * blockDim.x + threadIdx.x;
    if (idx >= B_ * HW_ * C_) return;
    int c = idx % C_;
    int i = (idx / C_) % HW_;
    int b = idx / (C_ * HW_);
    xt[idx] = (h16)x[((size_t)b * C_ + c) * HW_ + i];
}

// w [R][C][ks][ks] f32 -> wt [R][ks*ks][C] f16 (channel-contiguous A rows)
__global__ void mksa_cvt_wconv(const float* __restrict__ w, h16* __restrict__ wt,
                               int R, int ks) {
    int k2 = ks * ks;
    int n = R * k2 * C_;
    int idx = blockIdx.x * blockDim.x + threadIdx.x;
    if (idx >= n) return;
    int c = idx % C_;
    int p = (idx / C_) % k2;
    int r = idx / (C_ * k2);
    wt[idx] = (h16)w[((size_t)r * C_ + c) * k2 + p];
}

// plain f32 -> f16 (projection weights, already [co][ci] contiguous)
__global__ void mksa_cvt_w(const float* __restrict__ w, h16* __restrict__ wh, int n) {
    int idx = blockIdx.x * blockDim.x + threadIdx.x;
    if (idx < n) wh[idx] = (h16)w[idx];
}

// ---------------------------------------------------------------------------
// Implicit-GEMM convolution via WMMA (compile-time shape -> no idiv chains).
//   M = output channels (MTILES x 16), N = spatial (16 per wave), K = C per tap.
//   For each tap (ky,kx): 8 wmma steps over channels; per-lane zero pad OOB.
// OUT_MODE 0: out[((b*nbtot+slot)*HW + i)*RC + m]   (Q / K, i-major)
// OUT_MODE 1: out[(b*C + m)*HW + i]                 (V, c-major)
// ---------------------------------------------------------------------------
template <int KS, int MTILES, int OUT_MODE>
__global__ __launch_bounds__(256)
void mksa_conv_wmma(const h16* __restrict__ xt, const h16* __restrict__ wt,
                    h16* __restrict__ out, int slot, int nbtot, float scale) {
    constexpr int PAD = (KS - 1) / 2;
    constexpr int K2  = KS * KS;
    const int lane = threadIdx.x & 31;
    const int wave = threadIdx.x >> 5;
    const int gw   = blockIdx.x * 8 + wave;
    const int nt   = gw % NT_I;
    const int mt   = (gw / NT_I) % MTILES;
    const int b    = gw / (NT_I * MTILES);
    const int nIdx = lane & 15;
    const int hi   = lane >> 4;
    const int k0   = hi * 16;
    const int hsel = hi;
    const int i    = nt * 16 + nIdx;
    const int y    = i / Wd_, xc = i % Wd_;

    const h16* xbase = xt + (size_t)b * HW_ * C_;
    const h16* wbase = wt + (size_t)(mt * 16 + nIdx) * K2 * C_;
    v8f acc = {};
    for (int p = 0; p < K2; ++p) {
        const int ky = p / KS, kx = p % KS;
        const int yy = y + ky - PAD, xx = xc + kx - PAD;
        const bool inb = ((unsigned)yy < (unsigned)Hh_) && ((unsigned)xx < (unsigned)Wd_);
        const h16* xrow = xbase + (size_t)(yy * Wd_ + xx) * C_;
        const h16* wrow = wbase + (size_t)p * C_;
        #pragma unroll
        for (int kt = 0; kt < 8; ++kt) {
            const int c0 = kt * 32;
            v8h a1 = ((const v8h*)(wrow + c0))[2 * hsel];
            v8h a2 = ((const v8h*)(wrow + c0))[2 * hsel + 1];
            v16h av = VSHUF16(a1, a2);
            v16h bv = {};
            if (inb) bv = *(const v16h*)(xrow + c0 + k0);
            acc = __builtin_amdgcn_wmma_f32_16x16x32_f16(false, av, false, bv, (short)0, acc, false, false);
        }
    }
    #pragma unroll
    for (int v = 0; v < 8; ++v) {
        const int m = mt * 16 + v + 8 * hi;
        const float val = acc[v] * scale;
        if (OUT_MODE == 0)
            out[(((size_t)b * nbtot + slot) * HW_ + i) * RC_ + m] = (h16)val;
        else
            out[((size_t)b * C_ + m) * HW_ + i] = (h16)val;
    }
}

// ---------------------------------------------------------------------------
// Flash attention: one block per (i-tile, branch, batch); 8 waves.
//   wave 0:  S = Qt*K^T (2 wmma), online softmax -> P,alpha (ping-pong LDS)
//   all:     acc *= alpha; acc += P*V^T (2 wmma / wave, 32 V columns each)
// One barrier per 32-j step (ping-pong buffers); V staged via async-LDS.
// ---------------------------------------------------------------------------
__global__ __launch_bounds__(256)
void mksa_attn(const h16* __restrict__ Qt, const h16* __restrict__ Kt,
               const h16* __restrict__ Vh, h16* __restrict__ conc) {
    const int it   = blockIdx.x;
    const int br   = blockIdx.y;
    const int b    = blockIdx.z;
    const int lane = threadIdx.x & 31;
    const int wave = threadIdx.x >> 5;
    const int nIdx = lane & 15;
    const int hi   = lane >> 4;
    const int k0   = hi * 16;
    const int hsel = hi;
    const int i0   = it * 16;

    __shared__ __align__(64) h16 P_lds[2][16][32];
    __shared__ float alpha_lds[2][16];
    __shared__ float linv_lds[16];
#ifdef HAVE_ASYNC_LDS
    __shared__ __align__(64) h16 vst[2][8][32][32];  // [buf][wave][lane][2 frags x 16]
#endif

    float mrun[8], lrun[8];
    v16h aq = {};
    if (wave == 0) {
        #pragma unroll
        for (int v = 0; v < 8; ++v) { mrun[v] = -1e30f; lrun[v] = 0.f; }
        const h16* qrow = Qt + (((size_t)b * NB_ + br) * HW_ + (i0 + nIdx)) * RC_;
        v8h q1 = ((const v8h*)qrow)[2 * hsel];
        v8h q2 = ((const v8h*)qrow)[2 * hsel + 1];
        aq = VSHUF16(q1, q2);
    }

    const int  cBase = wave * 32;
    const h16* krow  = Kt + (size_t)b * HW_ * RC_;
    const h16* vrow0 = Vh + ((size_t)b * C_ + (cBase + nIdx)) * HW_;
    const h16* vrow1 = Vh + ((size_t)b * C_ + (cBase + 16 + nIdx)) * HW_;
    v8f acc0 = {}, acc1 = {};

#ifdef HAVE_ASYNC_LDS
    {   // prologue: stage V fragments for step 0 into buffer 0
        h16* dst = &vst[0][wave][lane][0];
        __builtin_amdgcn_global_load_async_to_lds_b128(AS1(vrow0 + k0),     AS3(dst),      0, 0);
        __builtin_amdgcn_global_load_async_to_lds_b128(AS1(vrow0 + k0 + 8), AS3(dst + 8),  0, 0);
        __builtin_amdgcn_global_load_async_to_lds_b128(AS1(vrow1 + k0),     AS3(dst + 16), 0, 0);
        __builtin_amdgcn_global_load_async_to_lds_b128(AS1(vrow1 + k0 + 8), AS3(dst + 24), 0, 0);
    }
#endif

    for (int jt = 0; jt < JST; ++jt) {
        const int j0 = jt * 32;
        const int p  = jt & 1;
        if (wave == 0) {
            const v16h bk0 = *(const v16h*)(krow + (size_t)(j0 + nIdx) * RC_ + k0);
            const v16h bk1 = *(const v16h*)(krow + (size_t)(j0 + 16 + nIdx) * RC_ + k0);
            if (jt + 1 < JST)
                __builtin_prefetch(krow + (size_t)(j0 + 32 + nIdx) * RC_ + k0, 0, 1);
            v8f zero = {};
            v8f s0 = __builtin_amdgcn_wmma_f32_16x16x32_f16(false, aq, false, bk0, (short)0, zero, false, false);
            v8f s1 = __builtin_amdgcn_wmma_f32_16x16x32_f16(false, aq, false, bk1, (short)0, zero, false, false);
            float rmax[8];
            #pragma unroll
            for (int v = 0; v < 8; ++v) rmax[v] = fmaxf(s0[v], s1[v]);
            #pragma unroll
            for (int m = 1; m < 16; m <<= 1) {
                #pragma unroll
                for (int v = 0; v < 8; ++v)
                    rmax[v] = fmaxf(rmax[v], __shfl_xor(rmax[v], m, 32));
            }
            float p0[8], p1[8], rsum[8], al[8];
            #pragma unroll
            for (int v = 0; v < 8; ++v) {
                float mn = fmaxf(mrun[v], rmax[v]);
                al[v]   = __expf(mrun[v] - mn);
                p0[v]   = __expf(s0[v] - mn);
                p1[v]   = __expf(s1[v] - mn);
                rsum[v] = p0[v] + p1[v];
                mrun[v] = mn;
            }
            #pragma unroll
            for (int m = 1; m < 16; m <<= 1) {
                #pragma unroll
                for (int v = 0; v < 8; ++v)
                    rsum[v] += __shfl_xor(rsum[v], m, 32);
            }
            #pragma unroll
            for (int v = 0; v < 8; ++v) {
                lrun[v] = al[v] * lrun[v] + rsum[v];
                int row = v + 8 * hi;
                P_lds[p][row][nIdx]      = (h16)p0[v];
                P_lds[p][row][16 + nIdx] = (h16)p1[v];
            }
            if (nIdx < 8) alpha_lds[p][nIdx + 8 * hi] = al[nIdx];
        }
        __syncthreads();

        // ---- all waves: rescale + accumulate P * V^T ----
        const v8h* prow = (const v8h*)(&P_lds[p][nIdx][0]);
        v8h pa = prow[2 * hsel];
        v8h pb = prow[2 * hsel + 1];
        v16h ap = VSHUF16(pa, pb);
#ifdef HAVE_ASYNC_LDS
        v16h bv0, bv1;
        if (jt + 1 < JST) {   // stage next step into other buffer, then consume current
            h16* dst = &vst[1 - p][wave][lane][0];
            const h16* s0g = vrow0 + j0 + 32 + k0;
            const h16* s1g = vrow1 + j0 + 32 + k0;
            __builtin_amdgcn_global_load_async_to_lds_b128(AS1(s0g),     AS3(dst),      0, 0);
            __builtin_amdgcn_global_load_async_to_lds_b128(AS1(s0g + 8), AS3(dst + 8),  0, 0);
            __builtin_amdgcn_global_load_async_to_lds_b128(AS1(s1g),     AS3(dst + 16), 0, 0);
            __builtin_amdgcn_global_load_async_to_lds_b128(AS1(s1g + 8), AS3(dst + 24), 0, 0);
            __builtin_amdgcn_s_wait_asynccnt(4);
        } else {
            __builtin_amdgcn_s_wait_asynccnt(0);
        }
        bv0 = *(const v16h*)(&vst[p][wave][lane][0]);
        bv1 = *(const v16h*)(&vst[p][wave][lane][16]);
#else
        const v16h bv0 = *(const v16h*)(vrow0 + j0 + k0);
        const v16h bv1 = *(const v16h*)(vrow1 + j0 + k0);
        if (jt + 1 < JST) {
            __builtin_prefetch(vrow0 + j0 + 32 + k0, 0, 1);
            __builtin_prefetch(vrow1 + j0 + 32 + k0, 0, 1);
        }
#endif
        float als[8];
        #pragma unroll
        for (int v = 0; v < 8; ++v) als[v] = alpha_lds[p][v + 8 * hi];
        #pragma unroll
        for (int v = 0; v < 8; ++v) { acc0[v] *= als[v]; acc1[v] *= als[v]; }
        acc0 = __builtin_amdgcn_wmma_f32_16x16x32_f16(false, ap, false, bv0, (short)0, acc0, false, false);
        acc1 = __builtin_amdgcn_wmma_f32_16x16x32_f16(false, ap, false, bv1, (short)0, acc1, false, false);
    }

    if (wave == 0) {
        #pragma unroll
        for (int v = 0; v < 8; ++v)
            if (nIdx == v) linv_lds[v + 8 * hi] = 1.0f / lrun[v];
    }
    __syncthreads();
    float li[8];
    #pragma unroll
    for (int v = 0; v < 8; ++v) li[v] = linv_lds[v + 8 * hi];
    #pragma unroll
    for (int v = 0; v < 8; ++v) {
        int i = i0 + v + 8 * hi;
        size_t base = ((size_t)b * HW_ + i) * FOURC + (size_t)br * C_;
        conc[base + cBase + nIdx]      = (h16)(acc0[v] * li[v]);
        conc[base + cBase + 16 + nIdx] = (h16)(acc1[v] * li[v]);
    }
}

// ---------------------------------------------------------------------------
// Projection GEMM (WMMA) + residual + deterministic BN partial sums.
// ---------------------------------------------------------------------------
__global__ __launch_bounds__(256)
void mksa_proj(const h16* __restrict__ wprojh, const h16* __restrict__ conc,
               const float* __restrict__ x, float* __restrict__ out_pre,
               float* __restrict__ psum, float* __restrict__ psq) {
    const int lane = threadIdx.x & 31;
    const int wave = threadIdx.x >> 5;
    const int gw   = blockIdx.x * 8 + wave;
    const int nt   = gw % NT_I;
    const int cot  = (gw / NT_I) % 16;
    const int b    = gw / (NT_I * 16);
    const int nIdx = lane & 15;
    const int hi   = lane >> 4;
    const int k0   = hi * 16;
    const int hsel = hi;
    const int i    = nt * 16 + nIdx;

    const h16* wrow = wprojh + (size_t)(cot * 16 + nIdx) * FOURC;
    const h16* crow = conc + ((size_t)b * HW_ + i) * FOURC;
    v8f acc = {};
    for (int kt = 0; kt < 32; ++kt) {
        const int ci0 = kt * 32;
        v8h a1 = ((const v8h*)(wrow + ci0))[2 * hsel];
        v8h a2 = ((const v8h*)(wrow + ci0))[2 * hsel + 1];
        v16h av = VSHUF16(a1, a2);
        v16h bv = *(const v16h*)(crow + ci0 + k0);
        if (kt + 1 < 32)
            __builtin_prefetch(crow + ci0 + 32 + k0, 0, 1);
        acc = __builtin_amdgcn_wmma_f32_16x16x32_f16(false, av, false, bv, (short)0, acc, false, false);
    }
    float rs[8], rq[8];
    #pragma unroll
    for (int v = 0; v < 8; ++v) {
        int co = cot * 16 + v + 8 * hi;
        size_t o = ((size_t)b * C_ + co) * HW_ + i;
        float val = acc[v] + x[o];
        out_pre[o] = val;
        rs[v] = val;
        rq[v] = val * val;
    }
    #pragma unroll
    for (int m = 1; m < 16; m <<= 1) {
        #pragma unroll
        for (int v = 0; v < 8; ++v) {
            rs[v] += __shfl_xor(rs[v], m, 32);
            rq[v] += __shfl_xor(rq[v], m, 32);
        }
    }
    #pragma unroll
    for (int v = 0; v < 8; ++v) {
        if (nIdx == v) {
            int co = cot * 16 + v + 8 * hi;
            psum[(size_t)co * NPART + b * NT_I + nt] = rs[v];
            psq [(size_t)co * NPART + b * NT_I + nt] = rq[v];
        }
    }
}

// per-channel BN stats (fixed-order tree reduction -> deterministic)
__global__ void mksa_bn_stats(const float* __restrict__ psum, const float* __restrict__ psq,
                              float* __restrict__ stats) {
    int c = blockIdx.x, t = threadIdx.x;
    __shared__ float ss[256], sq[256];
    float a = 0.f, q = 0.f;
    for (int j = t; j < NPART; j += 256) {
        a += psum[(size_t)c * NPART + j];
        q += psq [(size_t)c * NPART + j];
    }
    ss[t] = a; sq[t] = q;
    __syncthreads();
    for (int s = 128; s > 0; s >>= 1) {
        if (t < s) { ss[t] += ss[t + s]; sq[t] += sq[t + s]; }
        __syncthreads();
    }
    if (t == 0) {
        float n = (float)(B_ * HW_);
        float mean = ss[0] / n;
        float var  = sq[0] / n - mean * mean;
        stats[c]      = mean;
        stats[C_ + c] = rsqrtf(var + 1e-5f);
    }
}

__global__ void mksa_bn_apply(const float* __restrict__ out_pre, const float* __restrict__ stats,
                              const float* __restrict__ gamma, const float* __restrict__ beta,
                              float* __restrict__ out) {
    size_t idx = (size_t)blockIdx.x * blockDim.x + threadIdx.x;
    if (idx >= (size_t)B_ * C_ * HW_) return;
    int c = (int)((idx / HW_) % C_);
    out[idx] = (out_pre[idx] - stats[c]) * stats[C_ + c] * gamma[c] + beta[c];
}

// ---------------------------------------------------------------------------
extern "C" void kernel_launch(void* const* d_in, const int* in_sizes, int n_in,
                              void* d_out, int out_size, void* d_ws, size_t ws_size,
                              hipStream_t stream) {
    (void)in_sizes; (void)n_in; (void)out_size; (void)ws_size;
    const float* x     = (const float*)d_in[0];
    const float* wq1   = (const float*)d_in[1];
    const float* wq3   = (const float*)d_in[2];
    const float* wq5   = (const float*)d_in[3];
    const float* wq7   = (const float*)d_in[4];
    const float* wk    = (const float*)d_in[5];
    const float* wv    = (const float*)d_in[6];
    const float* wproj = (const float*)d_in[7];
    const float* gamma = (const float*)d_in[8];
    const float* beta  = (const float*)d_in[9];
    float* out = (float*)d_out;

    char* ws = (char*)d_ws;
    size_t off = 0;
    auto take = [&](size_t bytes) -> char* {
        off = (off + 255) & ~(size_t)255;
        char* p = ws + off;
        off += bytes;
        return p;
    };
    h16*   xt      = (h16*)  take((size_t)B_ * HW_ * C_ * sizeof(h16));
    h16*   Qt      = (h16*)  take((size_t)B_ * NB_ * HW_ * RC_ * sizeof(h16));
    h16*   Kt      = (h16*)  take((size_t)B_ * HW_ * RC_ * sizeof(h16));
    h16*   Vh      = (h16*)  take((size_t)B_ * C_ * HW_ * sizeof(h16));
    h16*   conc    = (h16*)  take((size_t)B_ * HW_ * FOURC * sizeof(h16));
    h16*   wprojh  = (h16*)  take((size_t)C_ * FOURC * sizeof(h16));
    h16*   wtq1    = (h16*)  take((size_t)RC_ * 1  * C_ * sizeof(h16));
    h16*   wtq3    = (h16*)  take((size_t)RC_ * 9  * C_ * sizeof(h16));
    h16*   wtq5    = (h16*)  take((size_t)RC_ * 25 * C_ * sizeof(h16));
    h16*   wtq7    = (h16*)  take((size_t)RC_ * 49 * C_ * sizeof(h16));
    h16*   wtk     = (h16*)  take((size_t)RC_ * 1  * C_ * sizeof(h16));
    h16*   wtv     = (h16*)  take((size_t)C_  * 1  * C_ * sizeof(h16));
    float* out_pre = (float*)take((size_t)B_ * C_ * HW_ * sizeof(float));
    float* psum    = (float*)take((size_t)C_ * NPART * sizeof(float));
    float* psq     = (float*)take((size_t)C_ * NPART * sizeof(float));
    float* stats   = (float*)take((size_t)2 * C_ * sizeof(float));

    const float qscale = 0.17677669529663689f;   // 1/sqrt(RC)

    // ---- layout transforms ----
    mksa_xt<<<(B_ * HW_ * C_ + 255) / 256, 256, 0, stream>>>(x, xt);
    mksa_cvt_w<<<(C_ * FOURC + 255) / 256, 256, 0, stream>>>(wproj, wprojh, C_ * FOURC);
    mksa_cvt_wconv<<<(RC_ * 1  * C_ + 255) / 256, 256, 0, stream>>>(wq1, wtq1, RC_, 1);
    mksa_cvt_wconv<<<(RC_ * 9  * C_ + 255) / 256, 256, 0, stream>>>(wq3, wtq3, RC_, 3);
    mksa_cvt_wconv<<<(RC_ * 25 * C_ + 255) / 256, 256, 0, stream>>>(wq5, wtq5, RC_, 5);
    mksa_cvt_wconv<<<(RC_ * 49 * C_ + 255) / 256, 256, 0, stream>>>(wq7, wtq7, RC_, 7);
    mksa_cvt_wconv<<<(RC_ * 1  * C_ + 255) / 256, 256, 0, stream>>>(wk,  wtk,  RC_, 1);
    mksa_cvt_wconv<<<(C_  * 1  * C_ + 255) / 256, 256, 0, stream>>>(wv,  wtv,  C_,  1);

    // ---- all convolutions as WMMA implicit GEMM ----
    const int gq = (B_ * NT_I * 2) / 8;     // q/k: mtiles=2 -> 196 blocks
    const int gv = (B_ * NT_I * 16) / 8;    // v:   mtiles=16 -> 1568 blocks
    mksa_conv_wmma<1, 2, 0><<<gq, 256, 0, stream>>>(xt, wtq1, Qt, 0, NB_, qscale);
    mksa_conv_wmma<3, 2, 0><<<gq, 256, 0, stream>>>(xt, wtq3, Qt, 1, NB_, qscale);
    mksa_conv_wmma<5, 2, 0><<<gq, 256, 0, stream>>>(xt, wtq5, Qt, 2, NB_, qscale);
    mksa_conv_wmma<7, 2, 0><<<gq, 256, 0, stream>>>(xt, wtq7, Qt, 3, NB_, qscale);
    mksa_conv_wmma<1, 2, 0><<<gq, 256, 0, stream>>>(xt, wtk,  Kt, 0, 1,   1.0f);
    mksa_conv_wmma<1, 16, 1><<<gv, 256, 0, stream>>>(xt, wtv, Vh, 0, 1,   1.0f);

    // ---- flash attention ----
    mksa_attn<<<dim3(NT_I, NB_, B_), 256, 0, stream>>>(Qt, Kt, Vh, conc);

    // ---- projection + residual + BN ----
    mksa_proj<<<(B_ * 16 * NT_I) / 8, 256, 0, stream>>>(wprojh, conc, x, out_pre, psum, psq);
    mksa_bn_stats<<<C_, 256, 0, stream>>>(psum, psq, stats);
    mksa_bn_apply<<<(B_ * C_ * HW_ + 255) / 256, 256, 0, stream>>>(out_pre, stats, gamma, beta, out);
}
